// AttentionBlock_67199058313653
// MI455X (gfx1250) — compile-verified, matrix operand order checked by reading, not run on hardware
//
#include <hip/hip_runtime.h>
#include <hip/hip_bf16.h>

typedef float v8f __attribute__((ext_vector_type(8)));
typedef __bf16 v16bf __attribute__((ext_vector_type(16)));
typedef unsigned int v8u __attribute__((ext_vector_type(8)));

#define DEVINL __device__ __forceinline__

constexpr int B_ = 8, C_ = 512, W_ = 128, HW_ = 9216;
constexpr int PW_ = 32, P_ = 576, D_ = 64;

// ---- fp32 -> packed bf16x2 conversion ----
// Prefer the single-instruction packed convert (v_cvt_pk_bf16_f32, RNE in HW).
// Fall back to inline asm of the same instruction if the builtin is not declared.
DEVINL unsigned short f2bf_raw(float f) {
  unsigned int u = __builtin_bit_cast(unsigned int, f);
  u += 0x7fffu + ((u >> 16) & 1u);
  return (unsigned short)(u >> 16);
}
DEVINL unsigned int cvt2bf(float lo, float hi) {
#if __has_builtin(__builtin_amdgcn_cvt_pk_bf16_f32)
  auto t = __builtin_amdgcn_cvt_pk_bf16_f32(lo, hi);
  return __builtin_bit_cast(unsigned int, t);
#else
  unsigned int r;
  asm("v_cvt_pk_bf16_f32 %0, %1, %2" : "=v"(r) : "v"(lo), "v"(hi));
  return r;
#endif
}
DEVINL v8f wmma_bf16(v16bf a, v16bf b, v8f c) {
  // (neg_a, A, neg_b, B, c_mod, C, reuse_a, reuse_b)
  return __builtin_amdgcn_wmma_f32_16x16x32_bf16(false, a, false, b, (short)0, c, false, false);
}
// per-lane K base for bf16 element pair j (pairs j=0..3 -> K kb+2j, j=4..7 -> K kb+8+2j)
DEVINL int kidx(int kb, int j) { return kb + ((j < 4) ? 2 * j : 8 + 2 * j); }

// ---------------- Kernel 1: exact 4x4 mean pool ----------------
__global__ void pool_kernel(const float* __restrict__ x, float* __restrict__ pooled) {
  int idx = blockIdx.x * 256 + threadIdx.x;       // over B*C*P
  int p = idx % P_;
  int bc = idx / P_;
  int ph = p / PW_, pw = p % PW_;
  const float* src = x + (size_t)bc * HW_ + (ph * 4) * W_ + pw * 4;
  float s = 0.f;
#pragma unroll
  for (int i = 0; i < 4; ++i) {
    float4 v = *(const float4*)(src + i * W_);
    s += (v.x + v.y) + (v.z + v.w);
  }
  pooled[idx] = s * 0.0625f;
}

// ---------------- Kernel 2: q|k = pooled @ [Q|K]  (M=4096, N=128, K=576) ----------------
__global__ void qk_kernel(const float* __restrict__ pooled,
                          const float* __restrict__ Qm, const float* __restrict__ Km,
                          float* __restrict__ qbuf, float* __restrict__ kbuf) {
  int lane = threadIdx.x & 31, wave = threadIdx.x >> 5;
  int mtile = blockIdx.x;                         // 256 tiles of 16 rows
  int col0 = wave * 16;                           // 0..127 (Q cols then K cols)
  const float* Bsrc = (col0 < 64) ? Qm : Km;
  int d0 = (col0 < 64) ? col0 : (col0 - 64);
  int m = mtile * 16 + (lane & 15);
  int n = d0 + (lane & 15);
  int khalf = (lane >> 4) << 3;
  v8f acc = 0.f;
  for (int ks = 0; ks < P_ / 32; ++ks) {
    int kb = ks * 32 + khalf;
    v8u au, bu;
#pragma unroll
    for (int j = 0; j < 8; ++j) {
      int kk = kidx(kb, j);
      au[j] = cvt2bf(pooled[(size_t)m * P_ + kk], pooled[(size_t)m * P_ + kk + 1]);
      bu[j] = cvt2bf(Bsrc[(size_t)kk * D_ + n], Bsrc[(size_t)(kk + 1) * D_ + n]);
    }
    acc = wmma_bf16(__builtin_bit_cast(v16bf, au), __builtin_bit_cast(v16bf, bu), acc);
  }
  float* dst = (col0 < 64) ? qbuf : kbuf;
#pragma unroll
  for (int r = 0; r < 8; ++r) {
    int row = mtile * 16 + r + khalf;
    dst[(size_t)row * D_ + n] = acc[r];
  }
}

// ------------- Kernel 3: a = q@k^T / sqrt(c), row softmax -> score  (per batch 512x512) -------------
__global__ void attn_kernel(const float* __restrict__ qbuf, const float* __restrict__ kbuf,
                            float* __restrict__ score) {
  __shared__ float s[16 * 512];                   // 32 KB: 16 rows of logits
  int lane = threadIdx.x & 31, wave = threadIdx.x >> 5;
  int mtile = blockIdx.x & 31;
  int b = blockIdx.x >> 5;
  int bC = b * C_;
  int khalf = (lane >> 4) << 3;
  int mrow = bC + mtile * 16 + (lane & 15);
  v8f acc[4];
#pragma unroll
  for (int t = 0; t < 4; ++t) acc[t] = 0.f;
#pragma unroll
  for (int ks = 0; ks < 2; ++ks) {                // K = 64
    int kb = ks * 32 + khalf;
    v8u au;
#pragma unroll
    for (int j = 0; j < 8; ++j) {
      int kk = kidx(kb, j);
      au[j] = cvt2bf(qbuf[(size_t)mrow * D_ + kk], qbuf[(size_t)mrow * D_ + kk + 1]);
    }
    v16bf a = __builtin_bit_cast(v16bf, au);
#pragma unroll
    for (int t = 0; t < 4; ++t) {                 // wave covers ntiles wave, wave+8, +16, +24
      int nrow = bC + (wave + t * 8) * 16 + (lane & 15);
      v8u bu;
#pragma unroll
      for (int j = 0; j < 8; ++j) {
        int kk = kidx(kb, j);
        bu[j] = cvt2bf(kbuf[(size_t)nrow * D_ + kk], kbuf[(size_t)nrow * D_ + kk + 1]);
      }
      acc[t] = wmma_bf16(a, __builtin_bit_cast(v16bf, bu), acc[t]);
    }
  }
  const float scale = 0.044194173824159216f;      // 1/sqrt(512)
#pragma unroll
  for (int t = 0; t < 4; ++t) {
    int col = (wave + t * 8) * 16 + (lane & 15);
#pragma unroll
    for (int r = 0; r < 8; ++r) s[(r + khalf) * 512 + col] = acc[t][r] * scale;
  }
  __syncthreads();
  // wave w owns rows 2w, 2w+1; wave32 shfl reductions
#pragma unroll
  for (int rr = 0; rr < 2; ++rr) {
    int row = wave * 2 + rr;
    float mx = -3.4e38f;
    for (int i = lane; i < 512; i += 32) mx = fmaxf(mx, s[row * 512 + i]);
#pragma unroll
    for (int off = 16; off > 0; off >>= 1) mx = fmaxf(mx, __shfl_xor(mx, off, 32));
    float sum = 0.f;
    for (int i = lane; i < 512; i += 32) {
      float e = __expf(s[row * 512 + i] - mx);
      s[row * 512 + i] = e;
      sum += e;
    }
#pragma unroll
    for (int off = 16; off > 0; off >>= 1) sum += __shfl_xor(sum, off, 32);
    float inv = 1.f / sum;
    size_t base = (size_t)(bC + mtile * 16 + row) * C_;
    for (int i = lane; i < 512; i += 32) score[base + i] = s[row * 512 + i] * inv;
  }
}

// ---------------- Kernel 4: bias_eff = score @ Vb ----------------
__global__ void beff_kernel(const float* __restrict__ score, const float* __restrict__ Vb,
                            float* __restrict__ beff) {
  int row = blockIdx.x * 256 + threadIdx.x;       // < B*C
  const float* sr = score + (size_t)row * C_;
  float acc = 0.f;
  for (int j = 0; j < C_; ++j) acc += sr[j] * Vb[j];
  beff[row] = acc;
}

// ---------------- Kernel 5: W_eff[b] = score[b] @ Vw  (512x512x512) ----------------
__global__ void weff_kernel(const float* __restrict__ score, const float* __restrict__ Vw,
                            float* __restrict__ weff) {
  int lane = threadIdx.x & 31, wave = threadIdx.x >> 5;
  int ng = blockIdx.x & 3;
  int mtile = (blockIdx.x >> 2) & 31;
  int b = blockIdx.x >> 7;
  int bC = b * C_;
  int col0 = (ng * 8 + wave) * 16;
  int mrow = bC + mtile * 16 + (lane & 15);
  int n = col0 + (lane & 15);
  int khalf = (lane >> 4) << 3;
  v8f acc = 0.f;
  for (int ks = 0; ks < C_ / 32; ++ks) {
    int kb = ks * 32 + khalf;
    v8u au, bu;
#pragma unroll
    for (int j = 0; j < 8; ++j) {
      int kk = kidx(kb, j);
      au[j] = cvt2bf(score[(size_t)mrow * C_ + kk], score[(size_t)mrow * C_ + kk + 1]);
      bu[j] = cvt2bf(Vw[(size_t)kk * C_ + n], Vw[(size_t)(kk + 1) * C_ + n]);
    }
    acc = wmma_bf16(__builtin_bit_cast(v16bf, au), __builtin_bit_cast(v16bf, bu), acc);
  }
#pragma unroll
  for (int r = 0; r < 8; ++r)
    weff[(size_t)(bC + mtile * 16 + r + khalf) * C_ + n] = acc[r];
}

// ---------------- Kernel 6: out[b] = W_eff[b] @ x[b] + bias_eff  (512x9216x512, memory-bound) ----------------
__global__ void out_kernel(const float* __restrict__ weff, const float* __restrict__ x,
                           const float* __restrict__ beff, float* __restrict__ out) {
  __shared__ __align__(16) unsigned short Abf[64][36];   // W_eff tile [m][k], bf16, padded row (72B, 8B-mult)
  __shared__ __align__(16) unsigned short Bt[128][36];   // x tile [n][k] (K-transposed), bf16, padded
  int tid = threadIdx.x;
  int lane = tid & 31, wave = tid >> 5;
  int nblk = blockIdx.x % 72;
  int mblk = (blockIdx.x / 72) & 7;
  int b = blockIdx.x / (72 * 8);
  int bC = b * C_;
  int m0 = mblk * 64, n0 = nblk * 128;
  int mi = wave >> 1;                              // 4 M-subtiles
  int nb = (wave & 1) * 4;                         // each wave: 4 of 8 N-subtiles
  int khalf = (lane >> 4) << 3;
  v8f acc[4];
#pragma unroll
  for (int t = 0; t < 4; ++t) acc[t] = 0.f;
  for (int ks = 0; ks < C_ / 32; ++ks) {
    int k0 = ks * 32;
    // Stage A: 64x32 fp32 -> bf16, B128 loads + B64 LDS stores
#pragma unroll
    for (int i = 0; i < 2; ++i) {
      int q = i * 256 + tid;                       // 512 float4 chunks
      int r = q >> 3, c4 = (q & 7) * 4;
      float4 v = *(const float4*)&weff[(size_t)(bC + m0 + r) * C_ + k0 + c4];
      uint2 u;
      u.x = cvt2bf(v.x, v.y);
      u.y = cvt2bf(v.z, v.w);
      *(uint2*)&Abf[r][c4] = u;
    }
    // Stage B: 32x128 fp32 -> bf16, B128 loads, transposed b16 LDS stores
#pragma unroll
    for (int i = 0; i < 4; ++i) {
      int q = i * 256 + tid;                       // 1024 float4 chunks
      int r = q >> 5, cc0 = (q & 31) * 4;
      float4 v = *(const float4*)&x[(size_t)(bC + k0 + r) * HW_ + n0 + cc0];
      unsigned int u01 = cvt2bf(v.x, v.y);
      unsigned int u23 = cvt2bf(v.z, v.w);
      Bt[cc0 + 0][r] = (unsigned short)u01;
      Bt[cc0 + 1][r] = (unsigned short)(u01 >> 16);
      Bt[cc0 + 2][r] = (unsigned short)u23;
      Bt[cc0 + 3][r] = (unsigned short)(u23 >> 16);
    }
    __syncthreads();
    if (ks + 1 < C_ / 32)                          // hint: global_prefetch next x tile
      __builtin_prefetch(&x[(size_t)(bC + k0 + 32 + (tid >> 7)) * HW_ + n0 + (tid & 127)], 0, 1);
    v8u au;
    const unsigned int* ap = (const unsigned int*)&Abf[mi * 16 + (lane & 15)][0];
#pragma unroll
    for (int j = 0; j < 8; ++j) au[j] = ap[kidx(khalf, j) >> 1];
    v16bf a = __builtin_bit_cast(v16bf, au);
#pragma unroll
    for (int t = 0; t < 4; ++t) {
      const unsigned int* bp = (const unsigned int*)&Bt[(nb + t) * 16 + (lane & 15)][0];
      v8u bu;
#pragma unroll
      for (int j = 0; j < 8; ++j) bu[j] = bp[kidx(khalf, j) >> 1];
      acc[t] = wmma_bf16(a, __builtin_bit_cast(v16bf, bu), acc[t]);
    }
    __syncthreads();
  }
#pragma unroll
  for (int t = 0; t < 4; ++t) {
    int col = n0 + (nb + t) * 16 + (lane & 15);
#pragma unroll
    for (int r = 0; r < 8; ++r) {
      int rowc = m0 + mi * 16 + r + khalf;
      out[(size_t)(bC + rowc) * HW_ + col] = acc[t][r] + beff[bC + rowc];
    }
  }
}

extern "C" void kernel_launch(void* const* d_in, const int* in_sizes, int n_in,
                              void* d_out, int out_size, void* d_ws, size_t ws_size,
                              hipStream_t stream) {
  (void)in_sizes; (void)n_in; (void)out_size; (void)ws_size;
  const float* x  = (const float*)d_in[0];
  const float* Vw = (const float*)d_in[1];
  const float* Vb = (const float*)d_in[2];
  const float* Qm = (const float*)d_in[3];
  const float* Km = (const float*)d_in[4];
  float* out = (float*)d_out;

  float* ws     = (float*)d_ws;                       // ~28.3 MB total
  float* pooled = ws;                                 // B*C*P
  float* qbuf   = pooled + (size_t)B_ * C_ * P_;      // B*C*D
  float* kbuf   = qbuf   + (size_t)B_ * C_ * D_;      // B*C*D
  float* score  = kbuf   + (size_t)B_ * C_ * D_;      // B*C*C
  float* weff   = score  + (size_t)B_ * C_ * C_;      // B*C*C
  float* beff   = weff   + (size_t)B_ * C_ * C_;      // B*C

  pool_kernel<<<(B_ * C_ * P_) / 256, 256, 0, stream>>>(x, pooled);
  qk_kernel<<<(B_ * C_) / 16, 256, 0, stream>>>(pooled, Qm, Km, qbuf, kbuf);
  attn_kernel<<<B_ * (C_ / 16), 256, 0, stream>>>(qbuf, kbuf, score);
  beff_kernel<<<(B_ * C_) / 256, 256, 0, stream>>>(score, Vb, beff);
  weff_kernel<<<B_ * (C_ / 16) * 4, 256, 0, stream>>>(score, Vw, weff);
  out_kernel<<<B_ * (C_ / 64) * (HW_ / 128), 256, 0, stream>>>(weff, x, beff, out);
}
